// BDeformConv_75574244540660
// MI455X (gfx1250) — matrix operand order repaired and checked by reading, not compiled, not run on hardware
//
#include <hip/hip_runtime.h>
#include <hip/hip_bf16.h>
#include <cstdint>

typedef __attribute__((ext_vector_type(16))) _Float16 v16h;
typedef __attribute__((ext_vector_type(8)))  float    v8f;
typedef __attribute__((ext_vector_type(2), aligned(4))) float v2fu; // 4B-aligned pair load

static constexpr int NB  = 8;     // batch
static constexpr int NC  = 64;    // input channels
static constexpr int NOC = 64;    // output channels
static constexpr int H   = 128;
static constexpr int W   = 128;
static constexpr int HW  = H * W;
static constexpr int K9  = 9;     // 3x3 taps
static constexpr int KTOT = NC * K9;      // 576 contraction length, idx = k9*64 + c
static constexpr int NKK  = KTOT / 32;    // 18 WMMA K-steps
static constexpr size_t FIELDS_OFF = 128 * 1024;  // apack (73728 B) lives below

// tanh/scale constants from reference
static constexpr float cA = 1.25f, cB = 1.75f, cCc = 1.0f, cD = 2.0f;

// ---------------------------------------------------------------------------
// Kernel 0: pre-swizzle w_main (64,64,3,3) into WMMA 16-bit A-matrix layout.
// apack[mt][kk][lane][j] ; K ordering idx = k9*64 + c.
// 16-bit A 16x32 (contiguous k order, matches B): lanes<16: K=j ; lanes>=16: K=j+16.
// ---------------------------------------------------------------------------
__global__ __launch_bounds__(256) void apack_kernel(const float* __restrict__ w_main,
                                                    _Float16* __restrict__ apack) {
    int t = blockIdx.x * blockDim.x + threadIdx.x;
    if (t >= 4 * NKK * 32) return;
    int mt   = t / (NKK * 32);
    int rem  = t - mt * (NKK * 32);
    int kk   = rem >> 5;
    int lane = rem & 31;
    int m    = mt * 16 + (lane & 15);
#pragma unroll
    for (int j = 0; j < 16; ++j) {
        int k_local = j + ((lane >= 16) ? 16 : 0);
        int kg = kk * 32 + k_local;          // 0..575
        int k9 = kg >> 6;                    // tap index
        int c  = kg & 63;                    // input channel
        apack[(size_t)t * 16 + j] = (_Float16)w_main[((size_t)m * NC + c) * K9 + k9];
    }
}

// ---------------------------------------------------------------------------
// Kernel 1: per-pixel fields (sin, cos, r, wr) via the three 3x3 convs.
// One block per (b,h) row, 128 threads = pixels. Weights staged in LDS as
// float4 per (c,tap): {w_rot0, w_rot1, w_str, w_whl}.
// ---------------------------------------------------------------------------
__global__ __launch_bounds__(128) void field_kernel(const float* __restrict__ x,
                                                    const float* __restrict__ w_rot,
                                                    const float* __restrict__ b_rot,
                                                    const float* __restrict__ w_str,
                                                    const float* __restrict__ b_str,
                                                    const float* __restrict__ w_whl,
                                                    const float* __restrict__ b_whl,
                                                    float4* __restrict__ fields) {
    __shared__ float4 wlds[NC * K9];
    float* wf = (float*)wlds;
    for (int i = threadIdx.x; i < NC * K9 * 4; i += blockDim.x) {
        int q = i & 3, ct = i >> 2;
        float v;
        if      (q == 0) v = w_rot[ct];
        else if (q == 1) v = w_rot[NC * K9 + ct];
        else if (q == 2) v = w_str[ct];
        else             v = w_whl[ct];
        wf[i] = v;
    }
    __syncthreads();

    int bh = blockIdx.x;
    int b = bh >> 7, h = bh & 127;
    int w = threadIdx.x;

    float as = 0.f, ac = 0.f, ar = 0.f, aw = 0.f;
    for (int c = 0; c < NC; ++c) {
        const float* xp = x + ((size_t)b * NC + c) * HW;
#pragma unroll
        for (int t = 0; t < K9; ++t) {
            int dy = t / 3 - 1, dx = t % 3 - 1;
            int yy = h + dy, xx = w + dx;
            float xv = (yy >= 0 && yy < H && xx >= 0 && xx < W) ? xp[(yy << 7) + xx] : 0.f;
            float4 wv = wlds[c * K9 + t];
            as += xv * wv.x; ac += xv * wv.y; ar += xv * wv.z; aw += xv * wv.w;
        }
    }
    float s  = as + b_rot[0];
    float co = ac + b_rot[1];
    float inv = rsqrtf(s * s + co * co + 1e-6f);
    s *= inv; co *= inv;
    float r  = tanhf(ar + b_str[0]) * cA + cB;
    float wr = tanhf(aw + b_whl[0]) * cCc + cD;
    fields[(bh << 7) + w] = make_float4(s, co, r, wr);
}

// ---------------------------------------------------------------------------
// Kernel 2: bilinear gather + WMMA contraction.
// One block per (b,h) row, 8 waves; wave i owns pixels [16i,16i+16) and all
// 64 output channels (4 M-tiles). K-dim idx = k9*64 + c -> each 32-wide
// K-step has a single k9 (= kk>>1) and c = ((kk&1)<<5) + 16*(lane>>4) + j.
//
// Tap table per (pixel n, k9): the two x-taps of each bilinear row are
// adjacent columns, so we load them as one b64 pair at base clamped to
// [0, W-2] and fold the column shift + validity masks into pair weights:
//   entry = { off_row0, off_row1, wa0, wb0, wa1, wb1, pad, pad }
//   val   = wa0*p[off0] + wb0*p[off0+1] + wa1*p[off1] + wb1*p[off1+1]
// ---------------------------------------------------------------------------
__global__ __launch_bounds__(256) void deform_wmma_kernel(const float* __restrict__ x,
                                                          const _Float16* __restrict__ apack,
                                                          const float4* __restrict__ fields,
                                                          float* __restrict__ out) {
    __shared__ __align__(16) unsigned int taps[8][16][K9][8];

    int tid  = threadIdx.x;
    int wave = tid >> 5;
    int lane = tid & 31;
    int bh   = blockIdx.x;
    int b = bh >> 7, h = bh & 127;
    int w0 = wave * 16;

    // ---- setup: per (pixel n, tap k9) pair offsets + folded weights ----
    for (int it = lane; it < 16 * K9; it += 32) {
        int n  = it / K9;
        int k9 = it - n * K9;
        float4 f = fields[(bh << 7) + w0 + n];   // s, co, r, wr
        float s = f.x, co = f.y, rr = f.z, ww = f.w;
        float ky = (float)(k9 / 3 - 1);
        float kx = (float)(k9 % 3 - 1);
        float o0 = ky * (ww * rr);
        float o1 = kx * ww;
        float py = (float)h        + co * o0 + s  * o1;
        float px = (float)(w0 + n) + co * o1 - s  * o0;
        float y0f = floorf(py), x0f = floorf(px);
        float ly = py - y0f, lx = px - x0f;
        int iy0 = (int)y0f, ix0 = (int)x0f;
        int iy1 = iy0 + 1;
        float vy0 = (iy0 >= 0 && iy0 < H) ? 1.f : 0.f;
        float vy1 = (iy1 >= 0 && iy1 < H) ? 1.f : 0.f;
        // column pair weights (shared by both rows), validity folded in
        float pa, pb;
        if (ix0 >= 0 && ix0 <= W - 2)      { pa = 1.f - lx; pb = lx; }
        else if (ix0 == -1)                { pa = lx;       pb = 0.f; }
        else if (ix0 == W - 1)             { pa = 0.f;      pb = 1.f - lx; }
        else                               { pa = 0.f;      pb = 0.f; }
        int cxa = min(max(ix0, 0), W - 2);
        int cy0 = min(max(iy0, 0), H - 1);
        int cy1 = min(max(iy1, 0), H - 1);
        float r0w = (1.f - ly) * vy0;
        float r1w = ly * vy1;
        unsigned int* tp = &taps[wave][n][k9][0];
        tp[0] = (unsigned)(cy0 * W + cxa);
        tp[1] = (unsigned)(cy1 * W + cxa);
        tp[2] = __float_as_uint(r0w * pa);
        tp[3] = __float_as_uint(r0w * pb);
        tp[4] = __float_as_uint(r1w * pa);
        tp[5] = __float_as_uint(r1w * pb);
        tp[6] = 0u;
        tp[7] = 0u;
    }
    __syncthreads();

    // ---- main loop: 18 K-steps, 4 WMMAs each ----
    v8f acc[4];
#pragma unroll
    for (int mt = 0; mt < 4; ++mt)
#pragma unroll
        for (int r = 0; r < 8; ++r) acc[mt][r] = 0.f;

    int n = lane & 15;
    int chi = (lane >> 4) << 4;                       // c contribution of lane half
    const float* xb = x + (size_t)b * NC * HW;

#pragma unroll 2
    for (int kk = 0; kk < NKK; ++kk) {
        int k9 = kk >> 1;
        const unsigned int* tp = &taps[wave][n][k9][0];
        uint4 t0 = *(const uint4*)(tp);               // off0, off1, wa0, wb0
        uint4 t1 = *(const uint4*)(tp + 4);           // wa1, wb1, 0, 0
        float wa0 = __uint_as_float(t0.z), wb0 = __uint_as_float(t0.w);
        float wa1 = __uint_as_float(t1.x), wb1 = __uint_as_float(t1.y);
        int cbase = ((kk & 1) << 5) + chi;
        const float* xc = xb + (size_t)cbase * HW;

        v16h bf;                                      // B[k][n]: lane = n+16*(k/16), half = k%16
#pragma unroll
        for (int j = 0; j < 16; ++j) {
            const float* p = xc + (size_t)j * HW;     // c = cbase + j
            v2fu r0 = *(const v2fu*)(p + t0.x);       // row y0: cols cxa, cxa+1
            v2fu r1 = *(const v2fu*)(p + t0.y);       // row y1: cols cxa, cxa+1
            float v = wa0 * r0.x + wb0 * r0.y + wa1 * r1.x + wb1 * r1.y;
            bf[j] = (_Float16)v;
        }
#pragma unroll
        for (int mt = 0; mt < 4; ++mt) {
            v16h af = *(const v16h*)(apack + (((size_t)(mt * NKK + kk) * 32 + lane) << 4));
            acc[mt] = __builtin_amdgcn_wmma_f32_16x16x32_f16(
                false, af, false, bf, (short)0, acc[mt], false, false);
        }
    }

    // ---- store D: VGPR r -> M = r + 8*(lane/16), N = lane%16 ----
    int wcol = w0 + n;
#pragma unroll
    for (int mt = 0; mt < 4; ++mt) {
#pragma unroll
        for (int r = 0; r < 8; ++r) {
            int oc = mt * 16 + (lane >> 4) * 8 + r;
            out[(((size_t)b * NOC + oc) * H + h) * W + wcol] = acc[mt][r];
        }
    }
}

// ---------------------------------------------------------------------------
extern "C" void kernel_launch(void* const* d_in, const int* in_sizes, int n_in,
                              void* d_out, int out_size, void* d_ws, size_t ws_size,
                              hipStream_t stream) {
    const float* x      = (const float*)d_in[0];
    const float* w_main = (const float*)d_in[1];
    const float* w_rot  = (const float*)d_in[2];
    const float* b_rot  = (const float*)d_in[3];
    const float* w_str  = (const float*)d_in[4];
    const float* b_str  = (const float*)d_in[5];
    const float* w_whl  = (const float*)d_in[6];
    const float* b_whl  = (const float*)d_in[7];
    float* out = (float*)d_out;

    _Float16* apack = (_Float16*)d_ws;                          // 4*18*32*16 f16 = 73728 B
    float4*   fields = (float4*)((char*)d_ws + FIELDS_OFF);     // 8*128*128 float4 = 2 MB

    apack_kernel<<<(4 * NKK * 32 + 255) / 256, 256, 0, stream>>>(w_main, apack);
    field_kernel<<<NB * H, 128, 0, stream>>>(x, w_rot, b_rot, w_str, b_str,
                                             w_whl, b_whl, fields);
    deform_wmma_kernel<<<NB * H, 256, 0, stream>>>(x, apack, fields, out);
}
